// MambaBlock_60636348285557
// MI455X (gfx1250) — matrix operand order repaired and checked by reading, not compiled, not run on hardware
//
#include <hip/hip_runtime.h>
#include <cstdint>
#include <cstddef>

// ---------------------------------------------------------------------------
// Mamba block for MI455X (gfx1250, wave32). fp32 I/O, f16 WMMA compute,
// async global->LDS staging, double-buffered tiles.
// ---------------------------------------------------------------------------

typedef __attribute__((ext_vector_type(16))) _Float16 v16h;
typedef __attribute__((ext_vector_type(8)))  _Float16 v8h;
typedef __attribute__((ext_vector_type(8)))  float    v8f;
typedef __attribute__((ext_vector_type(4)))  int      v4i;

constexpr int Bb   = 8;
constexpr int Ss   = 2048;
constexpr int Dm   = 512;     // model dim
constexpr int Di   = 1024;    // inner dim
constexpr int Nst  = 64;      // state dim
constexpr int Kc   = 4;       // conv width
constexpr int Rows = Bb * Ss; // 16384 GEMM rows

// ---- CDNA5 async global->LDS copy (guarded; falls back to direct copy) ----
#if defined(__has_builtin)
#  if __has_builtin(__builtin_amdgcn_global_load_async_to_lds_b128)
#    define HAVE_ASYNC_LDS 1
#  endif
#endif
#ifndef HAVE_ASYNC_LDS
#  define HAVE_ASYNC_LDS 0
#endif

typedef __attribute__((address_space(1))) v4i gv4i;   // global int4
typedef __attribute__((address_space(3))) v4i lv4i;   // LDS int4

__device__ __forceinline__ void cp_lds_b128(_Float16* l, const _Float16* g) {
#if HAVE_ASYNC_LDS
    __builtin_amdgcn_global_load_async_to_lds_b128(
        (gv4i*)(uintptr_t)g,
        (lv4i*)(uint32_t)(uintptr_t)l,
        0, 0);
#else
    *reinterpret_cast<v8h*>(l) = *reinterpret_cast<const v8h*>(g);
#endif
}

__device__ __forceinline__ void async_drain() {
#if HAVE_ASYNC_LDS
#  if defined(__has_builtin) && __has_builtin(__builtin_amdgcn_s_wait_asynccnt)
    __builtin_amdgcn_s_wait_asynccnt(0);
#  else
    asm volatile("s_wait_asynccnt 0" ::: "memory");
#  endif
#endif
}

__device__ __forceinline__ float sigmoid_f(float x) { return 1.0f / (1.0f + __expf(-x)); }
__device__ __forceinline__ float softplus_f(float x) {
    return (x > 20.0f) ? x : log1pf(__expf(x));
}

// ---------------------------------------------------------------------------
// One-time weight prep: w[K,N] fp32 -> wt[N,K] f16   (B operand, [col][k])
// ---------------------------------------------------------------------------
__global__ __launch_bounds__(256) void wprep_k(
    const float* __restrict__ w, _Float16* __restrict__ wt, int Kd, int Nd)
{
    const int idx = blockIdx.x * 256 + threadIdx.x;  // over Kd*Nd, coalesced read
    const int k = idx / Nd;
    const int n = idx - k * Nd;
    wt[(size_t)n * Kd + k] = (_Float16)w[idx];
}

// fp32 -> f16 elementwise
__global__ __launch_bounds__(256) void h16_k(
    const float* __restrict__ src, _Float16* __restrict__ dst)
{
    const int idx = blockIdx.x * 256 + threadIdx.x;
    dst[idx] = (_Float16)src[idx];
}

// ---------------------------------------------------------------------------
// Tiled WMMA GEMM:  C[M,N] = A[M,K] * Bt[N,K]^T   (A,Bt f16, fp32 accum)
// Block: 256 threads (8 waves). Block tile 128(M) x 128(N), K step 32,
// double-buffered async staging. Wave w: rows [w*16,w*16+16) x 128 cols.
// EPI: 0 none, 1 softplus(c + bias[col]), 2 c + resid[row*N+col]
// ---------------------------------------------------------------------------
template <int EPI>
__global__ __launch_bounds__(256) void gemm_wmma_k(
    const _Float16* __restrict__ A, const _Float16* __restrict__ Bt,
    float* __restrict__ C, int M, int Kd, int Nd,
    const float* __restrict__ bias, const float* __restrict__ resid)
{
    // 128 rows/cols x 32 halves, padded stride 40 halves (80 B) for banks.
    __shared__ _Float16 ldsA[2][128 * 40];
    __shared__ _Float16 ldsB[2][128 * 40];

    const int tid     = threadIdx.x;
    const int lane    = tid & 31;
    const int wave    = tid >> 5;
    const int rowbase = blockIdx.y * 128;
    const int nbase   = blockIdx.x * 128;
    const int lhalf   = (lane < 16);
    const int l16     = lane & 15;

    const _Float16* Ab  = A  + (size_t)rowbase * Kd;
    const _Float16* Bb_ = Bt + (size_t)nbase  * Kd;

    v8f acc[8];
#pragma unroll
    for (int nt = 0; nt < 8; ++nt) acc[nt] = (v8f)(0.0f);

    // stage tile kt into buffer buf: 512 x 16B chunks each for A and B
    auto stage = [&](int buf, int kbase) {
#pragma unroll
        for (int i = 0; i < 2; ++i) {
            const int idx = tid + i * 256;
            const int r = idx >> 2;       // 0..127
            const int q = idx & 3;        // 16B chunk in row
            cp_lds_b128(&ldsA[buf][r * 40 + q * 8], Ab  + (size_t)r * Kd + kbase + q * 8);
            cp_lds_b128(&ldsB[buf][r * 40 + q * 8], Bb_ + (size_t)r * Kd + kbase + q * 8);
        }
    };

    const int nkt = Kd / 32;
    stage(0, 0);
    async_drain();
    __syncthreads();

    for (int kt = 0; kt < nkt; ++kt) {
        const int buf = kt & 1;
        if (kt + 1 < nkt) stage(buf ^ 1, (kt + 1) * 32);   // overlap with compute

        // A fragment (16x32 f16 layout, ISA 7.12.2):
        // lanes 0-15: K 0-7 / 16-23 ; lanes 16-31: K 8-15 / 24-31
        const int m0 = wave * 16 + l16;
        const int ko = lhalf ? 0 : 8;
        const _Float16* la = &ldsA[buf][m0 * 40 + ko];
        v8h alo = *reinterpret_cast<const v8h*>(la);
        v8h ahi = *reinterpret_cast<const v8h*>(la + 16);
        v16h af;
#pragma unroll
        for (int j = 0; j < 8; ++j) { af[j] = alo[j]; af[j + 8] = ahi[j]; }

        // B fragments: lane holds column nbase+nt*16+l16, K 0-7 / 16-23 (lanes 0-15)
        // or K 8-15 / 24-31 (lanes 16-31), from the [col][k] LDS tile.
#pragma unroll
        for (int nt = 0; nt < 8; ++nt) {
            const _Float16* lb = &ldsB[buf][(nt * 16 + l16) * 40 + (lhalf ? 0 : 8)];
            v8h blo = *reinterpret_cast<const v8h*>(lb);
            v8h bhi = *reinterpret_cast<const v8h*>(lb + 16);
            v16h bf;
#pragma unroll
            for (int j = 0; j < 8; ++j) { bf[j] = blo[j]; bf[j + 8] = bhi[j]; }
            acc[nt] = __builtin_amdgcn_wmma_f32_16x16x32_f16(
                false, af, false, bf, (short)0, acc[nt], false, false);
        }

        async_drain();
        __syncthreads();
    }

    // C/D layout: VGPR r -> M = r (lanes 0-15) / r+8 (lanes 16-31), N = l16
#pragma unroll
    for (int nt = 0; nt < 8; ++nt) {
        const int col = nbase + nt * 16 + l16;
#pragma unroll
        for (int r = 0; r < 8; ++r) {
            const int row = rowbase + wave * 16 + r + (lhalf ? 0 : 8);
            float v = acc[nt][r];
            if constexpr (EPI == 1) v = softplus_f(v + bias[col]);
            if constexpr (EPI == 2) v += resid[(size_t)row * Nd + col];
            C[(size_t)row * Nd + col] = v;
        }
    }
}

// ---------------------------------------------------------------------------
// Causal depthwise conv (K=4) + bias + SiLU; also SiLU(z) and f16 copy of xc.
// ---------------------------------------------------------------------------
__global__ __launch_bounds__(256) void conv_silu_k(
    const float* __restrict__ xz, const float* __restrict__ w_conv,
    const float* __restrict__ b_conv, float* __restrict__ xc,
    _Float16* __restrict__ xch, float* __restrict__ silu_z)
{
    const int idx = blockIdx.x * 256 + threadIdx.x;   // over B*S*Di
    const int d = idx & (Di - 1);
    const int s = (idx / Di) & (Ss - 1);
    const int b = idx / (Di * Ss);

    float acc = b_conv[d];
#pragma unroll
    for (int k = 0; k < Kc; ++k) {
        const int sk = s - (Kc - 1) + k;
        if (sk >= 0)
            acc += w_conv[d * Kc + k] * xz[((size_t)b * Ss + sk) * (2 * Di) + d];
    }
    const float v = acc * sigmoid_f(acc);
    xc[(size_t)idx]  = v;
    xch[(size_t)idx] = (_Float16)v;

    const float zv = xz[((size_t)b * Ss + s) * (2 * Di) + Di + d];
    silu_z[(size_t)idx] = zv * sigmoid_f(zv);
}

// ---------------------------------------------------------------------------
// Selective scan: 256 channels/block of one batch, h[64]+A[64] in VGPRs,
// B_t/C_t double-buffered in LDS, one barrier per step.
// Emits yz = (y + D*x) * silu(z) directly as f16 (feeds the out-GEMM).
// ---------------------------------------------------------------------------
__global__ __launch_bounds__(256) void scan_k(
    const float* __restrict__ xc, const float* __restrict__ dtp,
    const float* __restrict__ xdbl, const float* __restrict__ A_log,
    const float* __restrict__ Dp, const float* __restrict__ sz,
    _Float16* __restrict__ yzh)
{
    const int tid = threadIdx.x;
    const int d   = blockIdx.x * 256 + tid;
    const int b   = blockIdx.y;

    __shared__ float bc[2][128];

    float h[Nst], a[Nst];
    const float Dv = Dp[d];
#pragma unroll
    for (int n = 0; n < Nst; ++n) {
        a[n] = -__expf(A_log[(size_t)d * Nst + n]);
        h[n] = 0.0f;
    }

    if (tid < 128) bc[0][tid] = xdbl[((size_t)b * Ss) * 128 + tid];
    __syncthreads();

    for (int t = 0; t < Ss; ++t) {
        const int buf = t & 1;
        if (t + 1 < Ss && tid < 128)
            bc[buf ^ 1][tid] = xdbl[((size_t)b * Ss + t + 1) * 128 + tid];

        const size_t idx = ((size_t)b * Ss + t) * Di + d;
        const float xv  = xc[idx];
        const float dtv = dtp[idx];
        const float dx  = dtv * xv;
        float y = Dv * xv;
#pragma unroll
        for (int n = 0; n < Nst; ++n) {
            const float ab = __expf(dtv * a[n]);   // v_exp_f32 co-executes w/ VALU
            h[n] = ab * h[n] + dx * bc[buf][n];
            y   += h[n] * bc[buf][64 + n];
        }
        yzh[idx] = (_Float16)(y * sz[idx]);
        __syncthreads();
    }
}

// ---------------------------------------------------------------------------
// LayerNorm over D=512 per row.
// ---------------------------------------------------------------------------
__global__ __launch_bounds__(256) void ln_k(
    const float* __restrict__ in, const float* __restrict__ g,
    const float* __restrict__ bta, float* __restrict__ out)
{
    const int row = blockIdx.x;
    const int tid = threadIdx.x;
    __shared__ float red[256];

    const float v0 = in[(size_t)row * Dm + tid];
    const float v1 = in[(size_t)row * Dm + 256 + tid];

    red[tid] = v0 + v1;
    __syncthreads();
    for (int s = 128; s > 0; s >>= 1) {
        if (tid < s) red[tid] += red[tid + s];
        __syncthreads();
    }
    const float mu = red[0] * (1.0f / Dm);
    __syncthreads();

    const float d0 = v0 - mu, d1 = v1 - mu;
    red[tid] = d0 * d0 + d1 * d1;
    __syncthreads();
    for (int s = 128; s > 0; s >>= 1) {
        if (tid < s) red[tid] += red[tid + s];
        __syncthreads();
    }
    const float rs = rsqrtf(red[0] * (1.0f / Dm) + 1e-5f);

    out[(size_t)row * Dm + tid]       = d0 * rs * g[tid]       + bta[tid];
    out[(size_t)row * Dm + 256 + tid] = d1 * rs * g[256 + tid] + bta[256 + tid];
}

// ---------------------------------------------------------------------------
extern "C" void kernel_launch(void* const* d_in, const int* in_sizes, int n_in,
                              void* d_out, int out_size, void* d_ws, size_t ws_size,
                              hipStream_t stream) {
    (void)in_sizes; (void)n_in; (void)out_size; (void)ws_size;
    const float* x      = (const float*)d_in[0];
    const float* w_in   = (const float*)d_in[1];
    const float* w_conv = (const float*)d_in[2];
    const float* b_conv = (const float*)d_in[3];
    const float* w_x    = (const float*)d_in[4];
    const float* w_dt   = (const float*)d_in[5];
    const float* b_dt   = (const float*)d_in[6];
    const float* A_log  = (const float*)d_in[7];
    const float* D_par  = (const float*)d_in[8];
    const float* w_out  = (const float*)d_in[9];
    const float* ln_g   = (const float*)d_in[10];
    const float* ln_b   = (const float*)d_in[11];
    float* out = (float*)d_out;

    // ---- workspace layout (bytes) ----
    const size_t R = (size_t)Rows;
    char* p = (char*)d_ws;
    float*    xz      = (float*)p;                 p += R * 2 * Di * 4;  // 134 MB
    float*    xcf     = (float*)p;                 p += R * Di * 4;      // 67 MB
    _Float16* xch     = (_Float16*)p;              p += R * Di * 2;      // 33.5 MB
    float*    szf     = (float*)p;                 p += R * Di * 4;      // 67 MB
    float*    xdbl    = (float*)p;                 p += R * 2 * Nst * 4; // 8.4 MB
    _Float16* xh      = (_Float16*)p;              p += R * Dm * 2;      // 16.8 MB
    _Float16* wt_in   = (_Float16*)p;              p += (size_t)Dm * 2 * Di * 2;
    _Float16* wt_x    = (_Float16*)p;              p += (size_t)Di * 2 * Nst * 2;
    _Float16* wt_dt   = (_Float16*)p;              p += (size_t)Di * Di * 2;
    _Float16* wt_out  = (_Float16*)p;              p += (size_t)Di * Dm * 2;
    // overlays on the dead xz region:
    float*    dt      = xz;                               // R*Di f32 (after conv)
    _Float16* yzh     = (_Float16*)(xz + R * Di);         // R*Di f16 (after conv)
    float*    out_pre = xz;                               // R*Dm f32 (after scan)

    // ---- 0) one-time operand prep (f16, B transposed to [N,K]) ----
    wprep_k<<<(Dm * 2 * Di) / 256, 256, 0, stream>>>(w_in,  wt_in,  Dm, 2 * Di);
    wprep_k<<<(Di * 2 * Nst) / 256, 256, 0, stream>>>(w_x,  wt_x,  Di, 2 * Nst);
    wprep_k<<<(Di * Di) / 256, 256, 0, stream>>>(w_dt, wt_dt, Di, Di);
    wprep_k<<<(Di * Dm) / 256, 256, 0, stream>>>(w_out, wt_out, Di, Dm);
    h16_k<<<(Rows * Dm) / 256, 256, 0, stream>>>(x, xh);

    // ---- 1) xz = x @ w_in            [16384,512]x[512,2048]
    gemm_wmma_k<0><<<dim3(2 * Di / 128, Rows / 128), 256, 0, stream>>>(
        xh, wt_in, xz, Rows, Dm, 2 * Di, nullptr, nullptr);

    // ---- 2) causal depthwise conv + SiLU ; SiLU(z)
    conv_silu_k<<<(Rows * Di) / 256, 256, 0, stream>>>(xz, w_conv, b_conv, xcf, xch, szf);

    // ---- 3) x_dbl = xc @ w_x         [16384,1024]x[1024,128]
    gemm_wmma_k<0><<<dim3(2 * Nst / 128, Rows / 128), 256, 0, stream>>>(
        xch, wt_x, xdbl, Rows, Di, 2 * Nst, nullptr, nullptr);

    // ---- 4) dt = softplus(xc @ w_dt + b_dt)
    gemm_wmma_k<1><<<dim3(Di / 128, Rows / 128), 256, 0, stream>>>(
        xch, wt_dt, dt, Rows, Di, Di, b_dt, nullptr);

    // ---- 5) selective scan -> yz (f16), fused *silu(z), +D*x
    scan_k<<<dim3(Di / 256, Bb), 256, 0, stream>>>(xcf, dt, xdbl, A_log, D_par, szf, yzh);

    // ---- 6) out_pre = yz @ w_out + residual(x)
    gemm_wmma_k<2><<<dim3(Dm / 128, Rows / 128), 256, 0, stream>>>(
        yzh, wt_out, out_pre, Rows, Di, Dm, nullptr, x);

    // ---- 7) LayerNorm
    ln_k<<<Rows, 256, 0, stream>>>(out_pre, ln_g, ln_b, out);
}
